// sRNN_69741678953279
// MI455X (gfx1250) — matrix operand order repaired and checked by reading, not compiled, not run on hardware
//
#include <hip/hip_runtime.h>

#define Bsz 64
#define Tt  1024
#define NXc 512
#define NUc 64
#define NYc 32

#if __has_builtin(__builtin_amdgcn_sched_barrier)
#define SCHED_BARRIER() __builtin_amdgcn_sched_barrier(0)
#else
#define SCHED_BARRIER()
#endif

typedef __attribute__((ext_vector_type(16))) __bf16 v16bf;
typedef __attribute__((ext_vector_type(8)))  float  v8f;

// A-fragment (16x32 MxK bf16, wave32) per-lane K mapping, per CDNA5 ISA 7.12.2:
// lane group g = lane>>4, element e (0..15): vgpr v=e>>1, pair p=e&1
// k = 16*(v>>2) + 2*(v&3) + p + 8*g
__device__ __forceinline__ int a_kmap(int g, int e) {
  int v = e >> 1, p = e & 1;
  return ((v >> 2) << 4) + ((v & 3) << 1) + p + (g << 3);
}
// Inverse: klocal (0..31) -> (group g2, element e)
__device__ __forceinline__ void a_kinv(int klocal, int& g2, int& e) {
  int hi  = klocal >> 4;
  int rem = klocal & 15;
  g2      = rem >> 3;
  int low = rem & 7;
  int v   = (hi << 2) | (low >> 1);
  e       = (v << 1) | (low & 1);
}

// ---------------------------------------------------------------------------
// Kernel 1: Gauss-Jordan inversion of E (512x512), one workgroup, aug=[E|I].
// Pivot row staged through LDS. E is well-conditioned near-identity (no pivoting).
// ---------------------------------------------------------------------------
__global__ __launch_bounds__(1024) void gj_inv_kernel(const float* __restrict__ E,
                                                      float* __restrict__ aug) {
  const int tid = threadIdx.x;
  for (int i = tid; i < 512 * 1024; i += 1024) {
    int r = i >> 10, c = i & 1023;
    aug[i] = (c < 512) ? E[r * 512 + c] : ((c - 512 == r) ? 1.0f : 0.0f);
  }
  __syncthreads();
  __shared__ float prow[1024];
  for (int p = 0; p < 512; ++p) {
    float pivinv = 1.0f / aug[p * 1024 + p];
    float rowval = aug[p * 1024 + tid] * pivinv;
    __syncthreads();
    aug[p * 1024 + tid] = rowval;
    prow[tid] = rowval;
    __syncthreads();
    int r = tid & 511, ch = tid >> 9;
    float f = aug[r * 1024 + p];
    __syncthreads();
    if (r != p) {
      int cbase = ch << 9;
      #pragma unroll 4
      for (int c = 0; c < 512; ++c)
        aug[r * 1024 + cbase + c] -= f * prow[cbase + c];
    }
    __syncthreads();
  }
}

// ---------------------------------------------------------------------------
// Kernel 2: pack an fp32 matrix into bf16 WMMA B-fragment layout.
// B fragment (32x16 KxN): lane = (n&15) + 16*((klocal>>3)&1), element e -> k = 8*(lane>>4)+e (e<8), 16+8*(lane>>4)+(e-8) (e>=8)
// dst[((kt*NNT+nt)*32 + lane)*16 + e], so each lane's 16 bf16 are contiguous (32B).
// ---------------------------------------------------------------------------
__global__ __launch_bounds__(256) void pack_b_kernel(const float* __restrict__ src,
                                                     int stride, int K, int N,
                                                     int transpose,
                                                     __bf16* __restrict__ dst) {
  int idx = blockIdx.x * blockDim.x + threadIdx.x;
  if (idx >= K * N) return;
  int k = idx / N, n = idx % N;
  float v = transpose ? src[n * stride + k] : src[k * stride + n];
  int NNT = N >> 4;
  int kt = k >> 5, klocal = k & 31, nt = n >> 4, nl = n & 15;
  int lane = nl + (((klocal >> 3) & 1) << 4);
  int e = (klocal & 7) + ((klocal >> 4) << 3);
  dst[(((size_t)kt * NNT + nt) * 32 + lane) * 16 + e] = (__bf16)v;
}

// ---------------------------------------------------------------------------
// Kernel 3: persistent WMMA scan. 4 workgroups (16 batch rows each), 16 waves.
// Wave w owns two N-tiles (columns 32w..32w+32); one LDS A-fragment feeds 2 WMMAs.
// A zero *offset* laundered through asm keeps loads non-loop-invariant but
// global-AS (global_load_b128, LOADcnt only). Depth-4 B-fragment ring + depth-2
// A-fragment ring, pinned with sched_barrier(0) so the post-RA scheduler cannot
// sink prefetches to their consumers -> partial s_wait_loadcnt waits, ~3 kt
// iterations of L2 slack per WMMA. GEMM2's ring is primed before the mid
// barrier so those loads fly during barrier wait.
// ---------------------------------------------------------------------------
__global__ __launch_bounds__(512) void scan_kernel(const float* __restrict__ inputs,
                                                   const float* __restrict__ h0,
                                                   const float* __restrict__ H_b,
                                                   const __bf16* __restrict__ HwT,
                                                   const __bf16* __restrict__ EinvT,
                                                   const __bf16* __restrict__ KwT,
                                                   float* __restrict__ states) {
  __shared__ __align__(32) __bf16 hA[16 * 32 * 16];  // 16 A-frags of h (bf16)
  __shared__ __align__(32) __bf16 xA[16 * 32 * 16];  // 16 A-frags of relu(x)
  __shared__ __align__(32) __bf16 uA[2 * 32 * 16];   // 2 A-frags of u_t

  const int tid   = threadIdx.x;
  const int lane  = tid & 31;
  const int w     = tid >> 5;          // wave id (0..15); owns N-tiles 2w, 2w+1
  const int bbase = blockIdx.x * 16;   // batch-row base
  const int grp   = lane >> 4;
  const int nl    = lane & 15;
  const int n0    = 2 * w, n1 = 2 * w + 1;

  // Prologue: h0 -> hA (A-layout bf16) and states[:,0,:] = h0 (fp32)
  for (int i = 0; i < 16; ++i) {
    int idx = i * 512 + tid;           // 0..8191
    int f = idx >> 9;
    int l = (idx >> 4) & 31;
    int e = idx & 15;
    int m = l & 15;
    int k = (f << 5) + a_kmap(l >> 4, e);
    hA[((f << 5) + l) * 16 + e] = (__bf16)h0[(bbase + m) * NXc + k];
  }
  for (int i = 0; i < 16; ++i) {
    int idx = i * 512 + tid;
    int m = idx >> 9;
    int x = idx & 511;
    states[((size_t)(bbase + m) * Tt) * NXc + x] = h0[(bbase + m) * NXc + x];
  }
  // u_0 -> uA
  for (int i = 0; i < 2; ++i) {
    int idx = i * 512 + tid;
    int f = idx >> 9, l = (idx >> 4) & 31, e = idx & 15;
    int m = l & 15;
    int k = (f << 5) + a_kmap(l >> 4, e);
    uA[((f << 5) + l) * 16 + e] = (__bf16)inputs[((size_t)(bbase + m) * Tt) * NUc + k];
  }

  // Hoisted loop-invariants: biases + 4 K_w fragments (~34 VGPRs)
  const float hb0 = H_b[(n0 << 4) + nl];
  const float hb1 = H_b[(n1 << 4) + nl];
  v16bf kw00 = ((const v16bf*)KwT)[(size_t)((0 * 32 + n0) * 32 + lane)];
  v16bf kw01 = ((const v16bf*)KwT)[(size_t)((0 * 32 + n1) * 32 + lane)];
  v16bf kw10 = ((const v16bf*)KwT)[(size_t)((1 * 32 + n0) * 32 + lane)];
  v16bf kw11 = ((const v16bf*)KwT)[(size_t)((1 * 32 + n1) * 32 + lane)];

  // D->A writeback: this wave's 32 columns are exactly K-fragment kt = w.
  // tile0 -> klocal = nl, tile1 -> klocal = 16 + nl
  int g2a, ea, g2b, eb;
  a_kinv(nl, g2a, ea);
  a_kinv(16 + nl, g2b, eb);
  const int fragBase = w << 5;  // lane base of frag w

  __syncthreads();

  for (int t = 0; t < Tt - 1; ++t) {
    // Launder a zero offset: breaks loop-invariance of the weight loads while
    // preserving the global address space of the base pointers.
    size_t zoff = 0;
    asm volatile("" : "+s"(zoff));
    const v16bf* HwF = (const v16bf*)(HwT + zoff);   // frag array [kt][nt][lane]
    const v16bf* EiF = (const v16bf*)(EinvT + zoff);

    // ---- GEMM1: x = h @ HwT + Hb + u @ KwT ----
    v8f c0 = {}, c1 = {};
    #pragma unroll
    for (int r = 0; r < 8; ++r) { c0[r] = hb0; c1[r] = hb1; }

    // Prime depth-4 B ring (stages 0..2) and depth-2 A ring (stage 0)
    v16bf rb0[4], rb1[4], ar[2];
    #pragma unroll
    for (int i = 0; i < 3; ++i) {
      rb0[i] = HwF[(size_t)(((i << 5) + n0) * 32 + lane)];
      rb1[i] = HwF[(size_t)(((i << 5) + n1) * 32 + lane)];
    }
    ar[0] = *(const v16bf*)&hA[(0 * 32 + lane) * 16];
    SCHED_BARRIER();
    // u contribution first (register-resident B) — covers ring fill latency
    {
      v16bf a0 = *(const v16bf*)&uA[(0 * 32 + lane) * 16];
      c0 = __builtin_amdgcn_wmma_f32_16x16x32_bf16(false, a0, false, kw00, (short)0, c0, false, false);
      c1 = __builtin_amdgcn_wmma_f32_16x16x32_bf16(false, a0, false, kw01, (short)0, c1, false, false);
      v16bf a1 = *(const v16bf*)&uA[(1 * 32 + lane) * 16];
      c0 = __builtin_amdgcn_wmma_f32_16x16x32_bf16(false, a1, false, kw10, (short)0, c0, false, false);
      c1 = __builtin_amdgcn_wmma_f32_16x16x32_bf16(false, a1, false, kw11, (short)0, c1, false, false);
    }
    SCHED_BARRIER();
    #pragma unroll
    for (int kt = 0; kt < 16; ++kt) {
      const int s = kt & 3;
      // prefetch B stage kt+3 (different ring slot; cannot sink past barrier)
      if (kt + 3 < 16) {
        const int sp = (kt + 3) & 3;
        rb0[sp] = HwF[(size_t)((((kt + 3) << 5) + n0) * 32 + lane)];
        rb1[sp] = HwF[(size_t)((((kt + 3) << 5) + n1) * 32 + lane)];
      }
      SCHED_BARRIER();
      // prefetch A stage kt+1 from LDS
      if (kt + 1 < 16)
        ar[(kt + 1) & 1] = *(const v16bf*)&hA[(((kt + 1) << 5) + lane) * 16];
      c0 = __builtin_amdgcn_wmma_f32_16x16x32_bf16(false, ar[kt & 1], false, rb0[s], (short)0, c0, false, false);
      c1 = __builtin_amdgcn_wmma_f32_16x16x32_bf16(false, ar[kt & 1], false, rb1[s], (short)0, c1, false, false);
      SCHED_BARRIER();
    }
    // relu -> xA (A-layout for GEMM2); wave writes all of frag kt=w
    #pragma unroll
    for (int r = 0; r < 8; ++r) {
      int m = r + (grp << 3);
      xA[((fragBase + (g2a << 4) + m) << 4) + ea] = (__bf16)fmaxf(c0[r], 0.0f);
      xA[((fragBase + (g2b << 4) + m) << 4) + eb] = (__bf16)fmaxf(c1[r], 0.0f);
    }
    // Prime GEMM2's B ring before the barrier: independent of xA, so these
    // loads are in flight while the workgroup barrier completes.
    v16bf sb0[4], sb1[4];
    #pragma unroll
    for (int i = 0; i < 3; ++i) {
      sb0[i] = EiF[(size_t)(((i << 5) + n0) * 32 + lane)];
      sb1[i] = EiF[(size_t)(((i << 5) + n1) * 32 + lane)];
    }
    SCHED_BARRIER();
    __syncthreads();

    // ---- GEMM2: h' = relu(x) @ EinvT ----
    v8f d0 = {}, d1 = {};
    v16bf ar2[2];
    ar2[0] = *(const v16bf*)&xA[(0 * 32 + lane) * 16];
    SCHED_BARRIER();
    #pragma unroll
    for (int kt = 0; kt < 16; ++kt) {
      const int s = kt & 3;
      if (kt + 3 < 16) {
        const int sp = (kt + 3) & 3;
        sb0[sp] = EiF[(size_t)((((kt + 3) << 5) + n0) * 32 + lane)];
        sb1[sp] = EiF[(size_t)((((kt + 3) << 5) + n1) * 32 + lane)];
      }
      SCHED_BARRIER();
      if (kt + 1 < 16)
        ar2[(kt + 1) & 1] = *(const v16bf*)&xA[(((kt + 1) << 5) + lane) * 16];
      d0 = __builtin_amdgcn_wmma_f32_16x16x32_bf16(false, ar2[kt & 1], false, sb0[s], (short)0, d0, false, false);
      d1 = __builtin_amdgcn_wmma_f32_16x16x32_bf16(false, ar2[kt & 1], false, sb1[s], (short)0, d1, false, false);
      SCHED_BARRIER();
    }
    // write states[:, t+1, :] (fp32) and hA (bf16 A-layout) for next step
    #pragma unroll
    for (int r = 0; r < 8; ++r) {
      int m = r + (grp << 3);
      size_t rowb = ((size_t)(bbase + m) * Tt + (t + 1)) * NXc;
      float v0 = d0[r], v1 = d1[r];
      states[rowb + (n0 << 4) + nl] = v0;
      states[rowb + (n1 << 4) + nl] = v1;
      hA[((fragBase + (g2a << 4) + m) << 4) + ea] = (__bf16)v0;
      hA[((fragBase + (g2b << 4) + m) << 4) + eb] = (__bf16)v1;
    }
    // prefetch u_{t+1} -> uA (all GEMM1 reads of uA completed before mid barrier)
    if (t < Tt - 2) {
      for (int i = 0; i < 2; ++i) {
        int idx = i * 512 + tid;
        int f = idx >> 9, l = (idx >> 4) & 31, e = idx & 15;
        int m = l & 15;
        int k = (f << 5) + a_kmap(l >> 4, e);
        uA[((f << 5) + l) * 16 + e] =
            (__bf16)inputs[((size_t)(bbase + m) * Tt + (t + 1)) * NUc + k];
      }
    }
    __syncthreads();
  }
}

// ---------------------------------------------------------------------------
// Kernel 4: yest = states @ W_out^T + b_out. (B*T,512)x(512,32) WMMA GEMM.
// 8 waves/block, 1 output 16x16 tile per wave.
// ---------------------------------------------------------------------------
__global__ __launch_bounds__(256) void yest_kernel(const float* __restrict__ states,
                                                   const __bf16* __restrict__ WoutT,
                                                   const float* __restrict__ b_out,
                                                   float* __restrict__ yest) {
  const int lane = threadIdx.x & 31;
  const int wave = threadIdx.x >> 5;
  const int tile = blockIdx.x * 8 + wave;  // 8192 tiles
  const int mt = tile >> 1, nt = tile & 1;
  const int grp = lane >> 4, nl = lane & 15;

  v8f c = {};
  const float bb = b_out[(nt << 4) + nl];
  #pragma unroll
  for (int r = 0; r < 8; ++r) c[r] = bb;

  const float* row = states + (size_t)((mt << 4) + nl) * NXc;
  #pragma unroll
  for (int kt = 0; kt < 16; ++kt) {
    v16bf a;
    #pragma unroll
    for (int v = 0; v < 8; ++v) {
      int k = (kt << 5) + ((v >> 2) << 4) + ((v & 3) << 1) + (grp << 3);
      a[2 * v]     = (__bf16)row[k];
      a[2 * v + 1] = (__bf16)row[k + 1];
    }
    v16bf b = ((const v16bf*)WoutT)[(size_t)(((kt << 1) + nt) * 32 + lane)];
    c = __builtin_amdgcn_wmma_f32_16x16x32_bf16(false, a, false, b, (short)0, c, false, false);
  }
  #pragma unroll
  for (int r = 0; r < 8; ++r) {
    int mm = r + (grp << 3);
    yest[(size_t)((mt << 4) + mm) * NYc + (nt << 4) + nl] = c[r];
  }
}

// ---------------------------------------------------------------------------
extern "C" void kernel_launch(void* const* d_in, const int* in_sizes, int n_in,
                              void* d_out, int out_size, void* d_ws, size_t ws_size,
                              hipStream_t stream) {
  const float* inputs = (const float*)d_in[0];  // [B,T,NU]
  const float* h0     = (const float*)d_in[1];  // [B,NX]
  const float* E      = (const float*)d_in[2];  // [NX,NX]
  const float* K_w    = (const float*)d_in[3];  // [NX,NU]
  const float* H_w    = (const float*)d_in[4];  // [NX,NX]
  const float* H_b    = (const float*)d_in[5];  // [NX]
  const float* W_out  = (const float*)d_in[6];  // [NY,NX]
  const float* b_out  = (const float*)d_in[7];  // [NY]

  float* yest   = (float*)d_out;                       // [B,T,NY]
  float* states = yest + (size_t)Bsz * Tt * NYc;       // [B,T,NX]

  // Workspace layout (~3.2 MB)
  char*   ws    = (char*)d_ws;
  float*  aug   = (float*)ws;                               // 512*1024 f32 (2 MB)
  __bf16* HwT   = (__bf16*)(ws + (size_t)512 * 1024 * 4);   // 512 KB
  __bf16* EinvT = HwT + (size_t)512 * 512;                  // 512 KB
  __bf16* KwT   = EinvT + (size_t)512 * 512;                // 64 KB
  __bf16* WoutT = KwT + (size_t)64 * 512;                   // 32 KB

  // 1) E_inv via Gauss-Jordan into right half of aug (row stride 1024)
  gj_inv_kernel<<<1, 1024, 0, stream>>>(E, aug);

  // 2) pack weights into bf16 WMMA B-fragment layouts
  const int PT = 256;
  pack_b_kernel<<<(512 * 512 + PT - 1) / PT, PT, 0, stream>>>(H_w, NXc, 512, 512, 1, HwT);
  pack_b_kernel<<<(512 * 512 + PT - 1) / PT, PT, 0, stream>>>(aug + 512, 1024, 512, 512, 0, EinvT);
  pack_b_kernel<<<(64 * 512 + PT - 1) / PT, PT, 0, stream>>>(K_w, NUc, 64, 512, 1, KwT);
  pack_b_kernel<<<(512 * 32 + PT - 1) / PT, PT, 0, stream>>>(W_out, NXc, 512, 32, 1, WoutT);

  // 3) sequential scan: 4 persistent workgroups x 16 waves (2 N-tiles per wave)
  scan_kernel<<<4, 512, 0, stream>>>(inputs, h0, H_b, HwT, EinvT, KwT, states);

  // 4) output projection: 8192 tiles / 8 waves per block
  yest_kernel<<<1024, 256, 0, stream>>>(states, WoutT, b_out, yest);
}
